// EGNN_9698036154372
// MI455X (gfx1250) — matrix operand order repaired and checked by reading, not compiled, not run on hardware
//
#include <hip/hip_runtime.h>
#include <hip/hip_bf16.h>

#define SD 128      // SDIM
#define HD 64       // HID
#define NDEPTH 4
#define PI_F 3.14159265358979323846f

typedef __attribute__((ext_vector_type(16))) __bf16 v16bf;
typedef __attribute__((ext_vector_type(2)))  __bf16 v2bf;
typedef __attribute__((ext_vector_type(8)))  float  v8f;

union Frag16 { v16bf v; __bf16 b[16]; unsigned short u[16]; };

// one-time pack kernel conversion (bit-exact RNE)
__device__ __forceinline__ unsigned short f2bf(float f) {
  unsigned u = __float_as_uint(f);
  unsigned r = u + 0x7FFFu + ((u >> 16) & 1u);
  return (unsigned short)(r >> 16);
}

// fast packed f32->bf16 for the hot path
__device__ __forceinline__ void pk2(__bf16* d, float x, float y) {
#if __has_builtin(__builtin_amdgcn_cvt_pk_bf16_f32)
  *(v2bf*)d = __builtin_amdgcn_cvt_pk_bf16_f32(x, y);
#else
  d[0] = (__bf16)x;
  d[1] = (__bf16)y;
#endif
}

__device__ __forceinline__ float silu_f(float x) {
#if __has_builtin(__builtin_amdgcn_rcpf)
  return x * __builtin_amdgcn_rcpf(1.0f + __expf(-x));
#else
  return x / (1.0f + __expf(-x));
#endif
}

__device__ __forceinline__ void dswait() {
  asm volatile("s_wait_dscnt 0" ::: "memory");
}

__device__ __forceinline__ v8f vzero8() {
  v8f z = {0.f,0.f,0.f,0.f,0.f,0.f,0.f,0.f};
  return z;
}

__device__ __forceinline__ v8f wmma_bf16(const Frag16& a, const Frag16& b, v8f c) {
  return __builtin_amdgcn_wmma_f32_16x16x32_bf16(
      false, a.v, false, b.v, (short)0, c, false, false);
}

// B fragment from pre-packed weights: 2 x b128 loads
__device__ __forceinline__ void load_bp(Frag16& b, const unsigned short* __restrict__ p,
                                        int t, int lane) {
  const uint4* q = (const uint4*)(p + ((t * 32 + lane) << 4));
  *(uint4*)&b.u[0] = q[0];
  *(uint4*)&b.u[8] = q[1];
}

// A fragment from LDS (bf16, row-major [row][K]): 2 x ds_load_b128
__device__ __forceinline__ void load_a_lds(Frag16& a, const __bf16* rowp, int kb) {
  *(uint4*)&a.u[0] = *(const uint4*)(rowp + kb);
  *(uint4*)&a.u[8] = *(const uint4*)(rowp + kb + 16);
}

// convert 8 f32 (two float4) into fragment elements [off..off+7]
__device__ __forceinline__ void cvt8(Frag16& a, int off, const float4 q0, const float4 q1) {
  pk2(a.b + off + 0, q0.x, q0.y);
  pk2(a.b + off + 2, q0.z, q0.w);
  pk2(a.b + off + 4, q1.x, q1.y);
  pk2(a.b + off + 6, q1.z, q1.w);
}

// ---------------------------------------------------------------------------
// Pack weight matrix (srcK x srcN, f32) into WMMA B-fragment order, bf16.
//   K = kt*32 + 8*(lane>=16) + (e<8 ? e : e+8) ; n = nt*16 + (lane&15), zero-padded n>=Nreal
// ---------------------------------------------------------------------------
__global__ void pack_b_kernel(const float* __restrict__ W, unsigned short* __restrict__ out,
                              int kts, int nts, int Nreal, int srcK, int srcN, long total) {
  long i = (long)blockIdx.x * blockDim.x + threadIdx.x;
  if (i >= total) return;
  int e2 = (int)(i & 15);
  long r = i >> 4;
  int lane = (int)(r & 31); r >>= 5;
  int nt = (int)(r % nts); r /= nts;
  int kt = (int)(r % kts); r /= kts;
  int mat = (int)r;
  int K = kt * 32 + ((e2 < 8) ? e2 : e2 + 8) + ((lane & 16) ? 8 : 0);
  int n = nt * 16 + (lane & 15);
  float val = (n < Nreal) ? W[((long)mat * srcK + K) * srcN + n] : 0.0f;
  out[i] = f2bf(val);
}

// ---------------------------------------------------------------------------
// Vectorized f32 -> bf16 shadow copy (8 elements / thread)
// ---------------------------------------------------------------------------
__global__ void cvt_s_kernel(const float* __restrict__ in,
                             unsigned short* __restrict__ out, long n8) {
  long i = (long)blockIdx.x * blockDim.x + threadIdx.x;
  if (i >= n8) return;
  const float4 q0 = *(const float4*)(in + i * 8);
  const float4 q1 = *(const float4*)(in + i * 8 + 4);
  union { __bf16 b[8]; uint4 q; } o;
  pk2(o.b + 0, q0.x, q0.y);
  pk2(o.b + 2, q0.z, q0.w);
  pk2(o.b + 4, q1.x, q1.y);
  pk2(o.b + 6, q1.z, q1.w);
  *(uint4*)(out + i * 8) = o.q;
}

// ---------------------------------------------------------------------------
// Precompute: C[e] = 0.5*(cos(pi*d/5)+1)*(d<5); deg[dst] += 1
// ---------------------------------------------------------------------------
__global__ void egnn_pre_kernel(const float* __restrict__ d,
                                const int* __restrict__ dst,
                                float* __restrict__ C, float* __restrict__ deg,
                                int E) {
  int e = blockIdx.x * blockDim.x + threadIdx.x;
  if (e < E) {
    float dd = d[e];
    float c = 0.5f * (__cosf(PI_F * dd * (1.0f / 5.0f)) + 1.0f);
    C[e] = (dd < 5.0f) ? c : 0.0f;
    atomicAdd(&deg[dst[e]], 1.0f);
  }
}

// ---------------------------------------------------------------------------
// Edge kernel: one wave32 per 16-edge tile. Gathers bf16 shadow of s.
// ---------------------------------------------------------------------------
#define EWAVES 4
__global__ __launch_bounds__(EWAVES * 32)
void egcl_edge_kernel(const unsigned short* __restrict__ sbf,
                      const float* __restrict__ vv,
                      const int* __restrict__ src, const int* __restrict__ dst,
                      const float* __restrict__ dq, const float* __restrict__ rr,
                      const float* __restrict__ Cc,
                      const unsigned short* __restrict__ We1p,
                      const float* __restrict__ We1_last,
                      const float* __restrict__ be1,
                      const unsigned short* __restrict__ We2p,
                      const float* __restrict__ be2,
                      const unsigned short* __restrict__ Wvp,
                      const float* __restrict__ bvp,
                      float* __restrict__ agg_s, float* __restrict__ agg_v,
                      int E) {
  __shared__ __align__(16) __bf16 lds_h[EWAVES][16 * HD];
  __shared__ float lds_g[EWAVES][16 * 8];

  const int lane = threadIdx.x & 31;
  const int wave = threadIdx.x >> 5;
  const int tile = blockIdx.x * EWAVES + wave;
  const int e0 = tile * 16;
  if (e0 >= E) return;                 // wave-uniform: EXEC stays all-ones for WMMA
  const int nrow = lane & 15;
  const int khalf = lane >> 4;

  __builtin_prefetch(We1p, 0, 0);

  const int erow = e0 + nrow;
  const unsigned short* __restrict__ srow = sbf + src[erow] * SD;
  const unsigned short* __restrict__ drow = sbf + dst[erow] * SD;

  // ---------------- GEMM1: (16 x 256) @ (256 x 64) ------------------------
  v8f acc[4] = {vzero8(), vzero8(), vzero8(), vzero8()};
#pragma unroll
  for (int kt = 0; kt < 8; ++kt) {
    Frag16 a;
    const int kb = kt * 32 + khalf * 8;   // elements 0..7 -> K=kb.. ; 8..15 -> K=kb+16..
    const int k1 = kb + 16;
    const unsigned short* p0 = (kb < SD) ? (drow + kb) : (srow + kb - SD);
    const unsigned short* p1 = (k1 < SD) ? (drow + k1) : (srow + k1 - SD);
    *(uint4*)&a.u[0] = *(const uint4*)p0;
    *(uint4*)&a.u[8] = *(const uint4*)p1;
#pragma unroll
    for (int nt = 0; nt < 4; ++nt) {
      Frag16 b;
      load_bp(b, We1p, kt * 4 + nt, lane);
      acc[nt] = wmma_bf16(a, b, acc[nt]);
    }
  }

  // vectorized side loads: d, dst, C for this lane's 8 rows
  const int mb = e0 + khalf * 8;
  const float4 dv0 = *(const float4*)(dq + mb);
  const float4 dv1 = *(const float4*)(dq + mb + 4);
  const int4 dm0 = *(const int4*)(dst + mb);
  const int4 dm1 = *(const int4*)(dst + mb + 4);
  const float4 cv0 = *(const float4*)(Cc + mb);
  const float4 cv1 = *(const float4*)(Cc + mb + 4);
  const float dv[8]  = {dv0.x, dv0.y, dv0.z, dv0.w, dv1.x, dv1.y, dv1.z, dv1.w};
  const int   dstm[8]= {dm0.x, dm0.y, dm0.z, dm0.w, dm1.x, dm1.y, dm1.z, dm1.w};
  const float cv[8]  = {cv0.x, cv0.y, cv0.z, cv0.w, cv1.x, cv1.y, cv1.z, cv1.w};

  // bias + fp32 rank-1 update (d column) + SiLU -> LDS (bf16)
#pragma unroll
  for (int nt = 0; nt < 4; ++nt) {
    const int nn = nt * 16 + nrow;
    const float bias = be1[nn];
    const float wl = We1_last[nn];
#pragma unroll
    for (int i = 0; i < 8; ++i) {
      int m = i + khalf * 8;
      float x = acc[nt][i] + bias + dv[i] * wl;
      lds_h[wave][m * HD + nn] = (__bf16)silu_f(x);
    }
  }
  dswait();

  // ---------------- GEMM2: (16 x 64) @ (64 x 64) --------------------------
  const __bf16* hrow = &lds_h[wave][nrow * HD];
  v8f acc2[4] = {vzero8(), vzero8(), vzero8(), vzero8()};
#pragma unroll
  for (int kt = 0; kt < 2; ++kt) {
    Frag16 a;
    load_a_lds(a, hrow, kt * 32 + khalf * 8);
#pragma unroll
    for (int nt = 0; nt < 4; ++nt) {
      Frag16 b;
      load_bp(b, We2p, kt * 4 + nt, lane);
      acc2[nt] = wmma_bf16(a, b, acc2[nt]);
    }
  }

  // h = silu(.)*C : scatter into agg_s (L2 atomics, int32 addressing) + restage
#pragma unroll
  for (int nt = 0; nt < 4; ++nt) {
    const int nn = nt * 16 + nrow;
    const float bias = be2[nn];
#pragma unroll
    for (int i = 0; i < 8; ++i) {
      int m = i + khalf * 8;
      float h = silu_f(acc2[nt][i] + bias) * cv[i];
      lds_h[wave][m * HD + nn] = (__bf16)h;
      atomicAdd(agg_s + ((dstm[i] << 6) + nn), h);
    }
  }
  dswait();

  // ---------------- gates = h @ Wv (64 x 6, N padded to 16) ----------------
  v8f accg = vzero8();
#pragma unroll
  for (int kt = 0; kt < 2; ++kt) {
    Frag16 a, b;
    load_a_lds(a, hrow, kt * 32 + khalf * 8);
    load_bp(b, Wvp, kt, lane);
    accg = wmma_bf16(a, b, accg);
  }
  if (nrow < 6) {
    const float bias = bvp[nrow];
#pragma unroll
    for (int i = 0; i < 8; ++i)
      lds_g[wave][(i + khalf * 8) * 8 + nrow] = accg[i] + bias;
  }
  dswait();

  // ---------------- vector message: one lane per edge ----------------------
  if (lane < 16) {
    const int e = e0 + lane;
    const int sn = src[e], dn = dst[e];
    float g1[3], g2[3];
#pragma unroll
    for (int j = 0; j < 3; ++j) {
      g1[j] = lds_g[wave][lane * 8 + j];
      g2[j] = lds_g[wave][lane * 8 + 3 + j];
    }
    const float* __restrict__ vsrc = vv + sn * 9;
    float* __restrict__ av = agg_v + dn * 9;
#pragma unroll
    for (int c = 0; c < 3; ++c) {
      float rc = rr[e * 3 + c];
#pragma unroll
      for (int j = 0; j < 3; ++j)
        atomicAdd(av + (c * 3 + j), vsrc[c * 3 + j] * g1[j] + rc * g2[j]);
    }
  }
}

// ---------------------------------------------------------------------------
// Node kernel: one wave32 per 16-node tile. Emits f32 s_out and bf16 shadow.
// ---------------------------------------------------------------------------
#define NWAVES 4
__global__ __launch_bounds__(NWAVES * 32)
void egcl_node_kernel(const float* __restrict__ s_in,
                      const unsigned short* __restrict__ sb_in,
                      const float* __restrict__ agg_s,
                      const float* __restrict__ v_in,
                      const float* __restrict__ agg_v,
                      const float* __restrict__ deg,
                      const unsigned short* __restrict__ Wn1p,
                      const float* __restrict__ bn1,
                      const unsigned short* __restrict__ Wn2p,
                      const float* __restrict__ bn2,
                      float* __restrict__ s_out,
                      unsigned short* __restrict__ sb_out,
                      float* __restrict__ v_out,
                      int N) {
  __shared__ __align__(16) __bf16 lds_u[NWAVES][16 * HD];
  __shared__ __align__(16) float  lds_o[NWAVES][16 * SD];

  const int lane = threadIdx.x & 31;
  const int wave = threadIdx.x >> 5;
  const int tile = blockIdx.x * NWAVES + wave;
  const int n0 = tile * 16;
  if (n0 >= N) return;
  const int nrow = lane & 15;
  const int khalf = lane >> 4;

  const int node = n0 + nrow;
  const unsigned short* __restrict__ sbrow = sb_in + node * SD;
  const float* __restrict__ arow = agg_s + node * HD;

  // GEMM1: (16 x 192) @ (192 x 64).  K<128 from bf16 shadow; K>=128 from f32 agg.
  v8f acc[4] = {vzero8(), vzero8(), vzero8(), vzero8()};
#pragma unroll
  for (int kt = 0; kt < 6; ++kt) {
    Frag16 a;
    const int kb = kt * 32 + khalf * 8;
    if (kt < 4) {
      *(uint4*)&a.u[0] = *(const uint4*)(sbrow + kb);
      *(uint4*)&a.u[8] = *(const uint4*)(sbrow + kb + 16);
    } else {
      const float* p0 = arow + (kb - SD);
      const float* p1 = arow + (kb - SD + 16);
      cvt8(a, 0, *(const float4*)p0, *(const float4*)(p0 + 4));
      cvt8(a, 8, *(const float4*)p1, *(const float4*)(p1 + 4));
    }
#pragma unroll
    for (int nt = 0; nt < 4; ++nt) {
      Frag16 b;
      load_bp(b, Wn1p, kt * 4 + nt, lane);
      acc[nt] = wmma_bf16(a, b, acc[nt]);
    }
  }
#pragma unroll
  for (int nt = 0; nt < 4; ++nt) {
    const int nn = nt * 16 + nrow;
    const float bias = bn1[nn];
#pragma unroll
    for (int i = 0; i < 8; ++i)
      lds_u[wave][(i + khalf * 8) * HD + nn] = (__bf16)silu_f(acc[nt][i] + bias);
  }
  dswait();

  // GEMM2: (16 x 64) @ (64 x 128), stage output tile in LDS
  const __bf16* urow = &lds_u[wave][nrow * HD];
  Frag16 a0, a1;
  load_a_lds(a0, urow, khalf * 8);
  load_a_lds(a1, urow, 32 + khalf * 8);
#pragma unroll
  for (int nt = 0; nt < 8; ++nt) {
    const int nn2 = nt * 16 + nrow;
    Frag16 b0, b1;
    load_bp(b0, Wn2p, 0 * 8 + nt, lane);
    load_bp(b1, Wn2p, 1 * 8 + nt, lane);
    v8f acc2 = vzero8();
    acc2 = wmma_bf16(a0, b0, acc2);
    acc2 = wmma_bf16(a1, b1, acc2);
    const float bias = bn2[nn2];
#pragma unroll
    for (int i = 0; i < 8; ++i)
      lds_o[wave][(i + khalf * 8) * SD + nn2] = acc2[i] + bias;
  }
  dswait();

  // coalesced residual writeout (f32 + bf16 shadow): lane covers cols [4l,4l+4)
#pragma unroll
  for (int t = 0; t < 16; ++t) {
    int gidx = (n0 + t) * SD + lane * 4;
    float4 resid = *(const float4*)(s_in + gidx);
    float4 val = *(const float4*)&lds_o[wave][t * SD + lane * 4];
    val.x += resid.x; val.y += resid.y; val.z += resid.z; val.w += resid.w;
    *(float4*)(s_out + gidx) = val;
    union { __bf16 b[4]; uint2 q; } ob;
    pk2(ob.b + 0, val.x, val.y);
    pk2(ob.b + 2, val.z, val.w);
    *(uint2*)(sb_out + gidx) = ob.q;
  }

  // vector residual: one lane per node
  if (lane < 16) {
    const int nd = n0 + lane;
    float dg = deg[nd];
    dg = (dg < 1.0f) ? 1.0f : dg;
    const float inv = 1.0f / dg;
#pragma unroll
    for (int t = 0; t < 9; ++t) {
      int idx = nd * 9 + t;
      v_out[idx] = v_in[idx] + agg_v[idx] * inv;
    }
  }
}

// ---------------------------------------------------------------------------
// Host launcher
// ---------------------------------------------------------------------------
extern "C" void kernel_launch(void* const* d_in, const int* in_sizes, int n_in,
                              void* d_out, int out_size, void* d_ws,
                              size_t ws_size, hipStream_t stream) {
  (void)n_in; (void)out_size; (void)ws_size;
  const float* s0   = (const float*)d_in[0];
  const float* v0   = (const float*)d_in[1];
  const int*   eidx = (const int*)d_in[2];
  const float* dd   = (const float*)d_in[3];
  const float* rr   = (const float*)d_in[4];
  const float* We1f = (const float*)d_in[6];
  const float* be1  = (const float*)d_in[7];
  const float* We2f = (const float*)d_in[8];
  const float* be2  = (const float*)d_in[9];
  const float* Wn1f = (const float*)d_in[10];
  const float* bn1  = (const float*)d_in[11];
  const float* Wn2f = (const float*)d_in[12];
  const float* bn2  = (const float*)d_in[13];
  const float* Wvf  = (const float*)d_in[14];
  const float* bvp  = (const float*)d_in[15];

  const int N = in_sizes[0] / SD;
  const int E = in_sizes[3];
  const int* src = eidx;
  const int* dstp = eidx + E;

  // ---- carve workspace (~99 MB) ----
  char* wp = (char*)d_ws;
  auto carve = [&](size_t bytes) -> char* {
    char* p = wp;
    wp += (bytes + 255) & ~(size_t)255;
    return p;
  };
  float* s_a  = (float*)carve((size_t)N * SD * 4);
  float* s_b  = (float*)carve((size_t)N * SD * 4);
  unsigned short* sbf_a = (unsigned short*)carve((size_t)N * SD * 2);
  unsigned short* sbf_b = (unsigned short*)carve((size_t)N * SD * 2);
  float* v_a  = (float*)carve((size_t)N * 9 * 4);
  float* v_b  = (float*)carve((size_t)N * 9 * 4);
  float* aggs = (float*)carve((size_t)N * HD * 4);
  float* aggv = (float*)carve((size_t)N * 9 * 4);
  float* degp = (float*)carve((size_t)N * 4);
  float* Cp   = (float*)carve((size_t)E * 4);
  // packed bf16 weights, fragment order: per-matrix sizes kts*nts*512
  unsigned short* We1p = (unsigned short*)carve((size_t)NDEPTH * 8 * 4 * 512 * 2);
  unsigned short* We2p = (unsigned short*)carve((size_t)NDEPTH * 2 * 4 * 512 * 2);
  unsigned short* Wn1p = (unsigned short*)carve((size_t)NDEPTH * 6 * 4 * 512 * 2);
  unsigned short* Wn2p = (unsigned short*)carve((size_t)NDEPTH * 2 * 8 * 512 * 2);
  unsigned short* Wvp  = (unsigned short*)carve((size_t)NDEPTH * 2 * 1 * 512 * 2);

  // ---- pack weights (once per launch) ----
  auto pack = [&](const float* W, unsigned short* out, int kts, int nts, int Nreal,
                  int srcK, int srcN) {
    long total = (long)NDEPTH * kts * nts * 512;
    pack_b_kernel<<<(unsigned)((total + 255) / 256), 256, 0, stream>>>(
        W, out, kts, nts, Nreal, srcK, srcN, total);
  };
  pack(We1f, We1p, 8, 4, 64, 257, 64);   // K rows 0..255; row 256 handled in fp32
  pack(We2f, We2p, 2, 4, 64, 64, 64);
  pack(Wn1f, Wn1p, 6, 4, 64, 192, 64);
  pack(Wn2f, Wn2p, 2, 8, 128, 64, 128);
  pack(Wvf,  Wvp,  2, 1, 6,  64, 6);

  // ---- bf16 shadow of the input s ----
  {
    long n8 = (long)N * SD / 8;
    cvt_s_kernel<<<(unsigned)((n8 + 255) / 256), 256, 0, stream>>>(s0, sbf_a, n8);
  }

  // ---- C and deg ----
  hipMemsetAsync(degp, 0, (size_t)N * 4, stream);
  egnn_pre_kernel<<<(E + 255) / 256, 256, 0, stream>>>(dd, dstp, Cp, degp, E);

  // ---- layers ----
  const float* s_cur = s0;
  const unsigned short* sb_cur = sbf_a;
  const float* v_cur = v0;
  float* outs = (float*)d_out;
  float* outv = outs + (size_t)N * SD;

  const int etiles = (E + 15) / 16;
  const int eblocks = (etiles + EWAVES - 1) / EWAVES;
  const int ntiles = (N + 15) / 16;
  const int nblocks = (ntiles + NWAVES - 1) / NWAVES;

  for (int layer = 0; layer < NDEPTH; ++layer) {
    hipMemsetAsync(aggs, 0, (size_t)N * HD * 4, stream);
    hipMemsetAsync(aggv, 0, (size_t)N * 9 * 4, stream);

    egcl_edge_kernel<<<eblocks, EWAVES * 32, 0, stream>>>(
        sb_cur, v_cur, src, dstp, dd, rr, Cp,
        We1p + (size_t)layer * 8 * 4 * 512,
        We1f + (size_t)layer * 257 * 64 + 256 * 64,   // fp32 d-column
        be1 + layer * 64,
        We2p + (size_t)layer * 2 * 4 * 512, be2 + layer * 64,
        Wvp + (size_t)layer * 2 * 1 * 512, bvp + layer * 6,
        aggs, aggv, E);

    float* s_nxt = (layer == NDEPTH - 1) ? outs : ((layer & 1) ? s_b : s_a);
    unsigned short* sb_nxt = (layer & 1) ? sbf_a : sbf_b;
    float* v_nxt = (layer == NDEPTH - 1) ? outv : ((layer & 1) ? v_b : v_a);

    egcl_node_kernel<<<nblocks, NWAVES * 32, 0, stream>>>(
        s_cur, sb_cur, aggs, v_cur, aggv, degp,
        Wn1p + (size_t)layer * 6 * 4 * 512, bn1 + layer * 64,
        Wn2p + (size_t)layer * 2 * 8 * 512, bn2 + layer * 128,
        s_nxt, sb_nxt, v_nxt, N);

    s_cur = s_nxt;
    sb_cur = sb_nxt;
    v_cur = v_nxt;
  }
}
